// AttentionBlock_90872918049266
// MI455X (gfx1250) — matrix operand order repaired and checked by reading, not compiled, not run on hardware
//
#include <hip/hip_runtime.h>
#include <hip/hip_bf16.h>

typedef unsigned short u16;
typedef __attribute__((ext_vector_type(16))) __bf16 v16bf;
typedef __attribute__((ext_vector_type(8)))  float  v8f;
typedef __attribute__((ext_vector_type(4)))  unsigned int u32x4;
typedef __attribute__((ext_vector_type(8)))  int         i32x8;
typedef __attribute__((ext_vector_type(4)))  int         i32x4;

// ---------------- constants (match reference) ----------------
static constexpr int  TT   = 2048;      // seq len
static constexpr int  BB   = 4;         // batch
static constexpr int  DD   = 512;       // model dim
static constexpr int  HH   = 8;         // heads
static constexpr int  HD   = 64;        // head dim
static constexpr int  FFN  = 2048;      // ffn dim
static constexpr int  NTOK = BB * TT;   // 8192 tokens

// ---------------- helpers ----------------
__device__ __forceinline__ u16 f2bf(float f) {
    unsigned int u = __float_as_uint(f);
    unsigned int r = u + 0x7fffu + ((u >> 16) & 1u);
    return (u16)(r >> 16);
}

__device__ __forceinline__ float gelu_exact(float v) {
    return 0.5f * v * (1.0f + erff(v * 0.70710678118654752f));
}

// ---------------- weight pack: fp32 [K,N] row-major -> bf16 WMMA B-fragment tiles ----
// Fragment layout (V_WMMA_*_16X16X32_BF16 B operand, 32x16):
//   lanes 0-15 : col = lane,    K = kt*32 + 0..7, 16..23
//   lanes 16-31: col = lane-16, K = kt*32 + 8..15, 24..31
// Each lane's 16 bf16 values stored contiguously (32B). For one ktile, 4 consecutive
// ntiles form a contiguous 4KB panel; two ktiles = one 2-row TDM transfer (8KB).
__global__ void __launch_bounds__(256)
pack_b_kernel(const float* __restrict__ src, u16* __restrict__ dst, int K, int N) {
    int idx   = blockIdx.x * blockDim.x + threadIdx.x;
    int total = (K >> 5) * (N >> 4) * 32;
    if (idx >= total) return;
    int lane = idx & 31;
    int tile = idx >> 5;
    int ntn  = N >> 4;
    int nt   = tile % ntn;
    int kt   = tile / ntn;
    int col  = nt * 16 + (lane & 15);
    int krow = kt * 32 + ((lane >> 4) << 3);
    u16* d = dst + (size_t)idx * 16;
#pragma unroll
    for (int e = 0; e < 16; ++e)
        d[e] = f2bf(src[(size_t)(krow + e + (e & 8)) * N + col]);
}

// ---------------- RMSNorm: fp32 row (D=512) -> bf16 row ----------------
__global__ void __launch_bounds__(256)
rmsnorm_kernel(const float* __restrict__ x, const float* __restrict__ w,
               u16* __restrict__ out) {
    int row = blockIdx.x;
    const float* xr = x + (size_t)row * DD;
    int t = threadIdx.x;                 // 256 threads, 2 elems each
    float a0 = xr[t], a1 = xr[t + 256];
    __shared__ float red[256];
    red[t] = a0 * a0 + a1 * a1;
    __syncthreads();
    for (int off = 128; off > 0; off >>= 1) {
        if (t < off) red[t] += red[t + off];
        __syncthreads();
    }
    float rinv = rsqrtf(red[0] / (float)DD + 1e-6f);
    out[(size_t)row * DD + t]       = f2bf(a0 * rinv * w[t]);
    out[(size_t)row * DD + t + 256] = f2bf(a1 * rinv * w[t + 256]);
}

// ---------------- WMMA fragment loads ----------------
// A (16x32 bf16): lane L<16 -> row L, K k0+0..7 & k0+16..23; lane L>=16 -> +8 K base
__device__ __forceinline__ v16bf loadA_frag(const u16* p) {
    union { uint4 q[2]; v16bf v; } u;
    u.q[0] = *(const uint4*)(p);
    u.q[1] = *(const uint4*)(p + 16);
    return u.v;
}
// B (32x16 bf16, pre-packed): lane reads 32 contiguous bytes (LDS)
__device__ __forceinline__ v16bf loadB_frag(const u16* p) {
    union { uint4 q[2]; v16bf v; } u;
    u.q[0] = *(const uint4*)(p);
    u.q[1] = *(const uint4*)(p + 8);
    return u.v;
}

// ---------------- TDM: copy `rows` x 4KB B panels to LDS (contiguous dest) ----------
// D# per CDNA5 ISA: group0 = {count=1, lds_addr, global_addr,
// type=2}; group1 = {data_size=4B, tensor_dim0=1024, tensor_dim1=rows, tile_dim0=1024,
// tile_dim1=rows, tensor_dim0_stride=stride_dw}; groups 2/3 + extension zero (<=2D).
__device__ __forceinline__ void tdm_load_panel(const u16* gptr, unsigned lds_off,
                                               unsigned rows, unsigned stride_dw) {
    unsigned long long ga = (unsigned long long)(uintptr_t)gptr;
    u32x4 g0;
    g0[0] = 1u;                                            // count = 1 (valid D#)
    g0[1] = lds_off;                                       // LDS byte address
    g0[2] = (unsigned)(ga & 0xFFFFFFFFu);                  // global addr [31:0]
    g0[3] = (unsigned)((ga >> 32) & 0x01FFFFFFu) | (2u << 30); // addr[56:32] | type=2
    i32x8 g1;
    g1[0] = (int)(2u << 16);                               // data_size = 2 -> 4 bytes
    g1[1] = (int)(1024u << 16);                            // tensor_dim0[15:0]
    g1[2] = (int)(rows << 16);                             // td0 hi | tensor_dim1 lo16
    g1[3] = (int)(1024u << 16);                            // td1 hi | tile_dim0 = 1024
    g1[4] = (int)rows;                                     // tile_dim1 | tile_dim2 = 0
    g1[5] = (int)stride_dw;                                // tensor_dim0_stride lo32
    g1[6] = 0;
    g1[7] = 0;
    i32x4 z4 = {0, 0, 0, 0};
    i32x8 z8 = {0, 0, 0, 0, 0, 0, 0, 0};
    __builtin_amdgcn_tensor_load_to_lds(g0, g1, z4, z4, z8, 0);
}

// A fragments for one 64-K stage: 2 m-tiles x 2 ktiles
struct AF { v16bf a00, a10, a01, a11; };   // aMK: m-tile M, ktile K-within-stage

__device__ __forceinline__ void load_stageA(AF& f, const u16* aptr0, const u16* aptr1,
                                            int kbase) {
    f.a00 = loadA_frag(aptr0 + kbase);
    f.a10 = loadA_frag(aptr1 + kbase);
    f.a01 = loadA_frag(aptr0 + kbase + 32);
    f.a11 = loadA_frag(aptr1 + kbase + 32);
}

#define WMMA_BF16(a, b, c) \
    __builtin_amdgcn_wmma_f32_16x16x32_bf16(false, (a), false, (b), (short)0, (c), false, false)

__device__ __forceinline__ void stage_mma(v8f acc[8], const AF& f,
                                          const u16* buf, int lane) {
    const u16* p = buf + (size_t)lane * 16;
    {   // ktile 0 of stage
        v16bf b0 = loadB_frag(p);
        v16bf b1 = loadB_frag(p + 512);
        v16bf b2 = loadB_frag(p + 1024);
        v16bf b3 = loadB_frag(p + 1536);
        acc[0] = WMMA_BF16(f.a00, b0, acc[0]);
        acc[1] = WMMA_BF16(f.a00, b1, acc[1]);
        acc[2] = WMMA_BF16(f.a00, b2, acc[2]);
        acc[3] = WMMA_BF16(f.a00, b3, acc[3]);
        acc[4] = WMMA_BF16(f.a10, b0, acc[4]);
        acc[5] = WMMA_BF16(f.a10, b1, acc[5]);
        acc[6] = WMMA_BF16(f.a10, b2, acc[6]);
        acc[7] = WMMA_BF16(f.a10, b3, acc[7]);
    }
    {   // ktile 1 of stage (second 4KB row in LDS)
        v16bf b0 = loadB_frag(p + 2048);
        v16bf b1 = loadB_frag(p + 2560);
        v16bf b2 = loadB_frag(p + 3072);
        v16bf b3 = loadB_frag(p + 3584);
        acc[0] = WMMA_BF16(f.a01, b0, acc[0]);
        acc[1] = WMMA_BF16(f.a01, b1, acc[1]);
        acc[2] = WMMA_BF16(f.a01, b2, acc[2]);
        acc[3] = WMMA_BF16(f.a01, b3, acc[3]);
        acc[4] = WMMA_BF16(f.a11, b0, acc[4]);
        acc[5] = WMMA_BF16(f.a11, b1, acc[5]);
        acc[6] = WMMA_BF16(f.a11, b2, acc[6]);
        acc[7] = WMMA_BF16(f.a11, b3, acc[7]);
    }
}

// One 64-K stage: wait current TDM, barrier (publish + retire-guarantee), kick next
// TDM into the other LDS buffer, register-pipeline next stage's A, then 16 WMMAs.
__device__ __forceinline__ void gemm_stage(int st, int nst, int wid, int lane,
        const u16* aptr0, const u16* aptr1, const u16* bpanel, size_t bstrideK,
        unsigned stride_dw, const u16* bufcur, unsigned ldsNext,
        AF& cur, AF& nxt, v8f acc[8]) {
    if (wid == 0) __builtin_amdgcn_s_wait_tensorcnt(0);   // stage st staged in LDS
    __syncthreads();                                       // single barrier per stage
    if (wid == 0 && st + 1 < nst)                          // stage st+1 (async, other buf)
        tdm_load_panel(bpanel + (size_t)(2 * (st + 1)) * bstrideK, ldsNext, 2, stride_dw);
    if (st + 1 < nst) load_stageA(nxt, aptr0, aptr1, (st + 1) * 64);
    if (st + 2 < nst) {
        __builtin_prefetch(aptr0 + (st + 2) * 64, 0, 1);
        __builtin_prefetch(aptr1 + (st + 2) * 64, 0, 1);
    }
    stage_mma(acc, cur, bufcur, lane);
}

// ---------------- WMMA GEMM: C[M,N] = A[M,K](bf16,row-major) x Bpacked + epilogue ----
// 8 waves/block; each wave owns a 32x64 strip. B staged in LDS by the Tensor Data
// Mover (8KB per 64-K stage, double buffered); manual unroll x2 over stages gives
// ping-pong A registers with no copies. __launch_bounds__(256) lifts the default
// 1024-thread VGPR cap (128/wave) that was forcing A-fragment spills to scratch.
// Requires K % 128 == 0 (call sites: 512, 2048).
__global__ void __launch_bounds__(256)
gemm_bf16_wmma(const u16* __restrict__ A, const u16* __restrict__ Bp,
               const float* __restrict__ bias,
               const float* __restrict__ resid,
               float* __restrict__ Cf, u16* __restrict__ Cbf,
               int M, int N, int K, int gelu_flag) {
    __shared__ __align__(16) u16 bstage[2][4096];          // 2 x 8KB double buffer

    int wid  = threadIdx.x >> 5;
    int lane = threadIdx.x & 31;
    int rowBase = (blockIdx.y * 8 + wid) << 5;             // 32 rows per wave
    int nt0  = blockIdx.x * 4;                             // 4 x 16 cols
    int ntn  = N >> 4;
    int nst  = K >> 6;                                     // 64-K stages (even)

    v8f acc[8];
#pragma unroll
    for (int s = 0; s < 8; ++s)
#pragma unroll
        for (int i = 0; i < 8; ++i) acc[s][i] = 0.0f;

    const u16* aptr0 = A + (size_t)(rowBase + (lane & 15)) * K + ((lane >> 4) << 3);
    const u16* aptr1 = aptr0 + (size_t)16 * K;
    size_t bstrideK   = (size_t)ntn * 512;                 // ushorts per 32-K tile
    unsigned strideDw = (unsigned)ntn * 256;               // dwords per 32-K tile
    const u16* bpanel = Bp + (size_t)nt0 * 512;            // this block's panel column

    unsigned lds0 = (unsigned)(uintptr_t)&bstage[0][0];    // LDS byte offsets
    unsigned lds1 = (unsigned)(uintptr_t)&bstage[1][0];

    // prologue: stage 0 via TDM, A fragments for stage 0
    if (wid == 0) tdm_load_panel(bpanel, lds0, 2, strideDw);
    AF aE, aO;
    load_stageA(aE, aptr0, aptr1, 0);
    aO = aE;

    for (int st = 0; st < nst; st += 2) {
        gemm_stage(st,     nst, wid, lane, aptr0, aptr1, bpanel, bstrideK, strideDw,
                   &bstage[0][0], lds1, aE, aO, acc);
        gemm_stage(st + 1, nst, wid, lane, aptr0, aptr1, bpanel, bstrideK, strideDw,
                   &bstage[1][0], lds0, aO, aE, acc);
    }

    int rsel = (lane >> 4) << 3;   // 0 or 8
#pragma unroll
    for (int m = 0; m < 2; ++m) {
        int rb = rowBase + (m << 4) + rsel;
#pragma unroll
        for (int s = 0; s < 4; ++s) {
            int col  = ((nt0 + s) << 4) + (lane & 15);
            float bv = bias ? bias[col] : 0.0f;
#pragma unroll
            for (int v = 0; v < 8; ++v) {
                int row = rb + v;
                float val = acc[m * 4 + s][v] + bv;
                if (gelu_flag) val = gelu_exact(val);
                if (resid) val += resid[(size_t)row * N + col];
                if (Cf)  Cf [(size_t)row * N + col] = val;
                if (Cbf) Cbf[(size_t)row * N + col] = f2bf(val);
            }
        }
    }
}

// ---------------- banded attention: one wave per (b,h,q) ----------------
// qkv fp32 [NTOK, 3*DD]; out bf16 [NTOK, DD]
__global__ void __launch_bounds__(256)
attn_band_kernel(const float* __restrict__ qkv,
                 u16* __restrict__ out_bf,
                 const int* __restrict__ ctx_ptr) {
    __shared__ float sc[8][132];
    int ctx  = *ctx_ptr;
    int wid  = threadIdx.x >> 5;
    int lane = threadIdx.x & 31;
    int gw   = blockIdx.x * 8 + wid;     // query index
    int b    = gw >> 14;                 // H*T = 16384
    int rem  = gw & 16383;
    int h    = rem >> 11;                // T = 2048
    int t    = rem & 2047;
    int bT   = b * TT;
    int row  = bT + t;

    int j0 = t - ctx; if (j0 < 0) j0 = 0;
    int j1 = t + ctx; if (j1 > TT - 1) j1 = TT - 1;
    int nk = j1 - j0 + 1;                // <= 129

    // cache Q (64 floats) in registers
    const float4* qp = (const float4*)(qkv + (size_t)row * (3 * DD) + h * HD);
    float4 qv[16];
#pragma unroll
    for (int i = 0; i < 16; ++i) qv[i] = qp[i];

    // pass 1: scores
    for (int jj = lane; jj < nk; jj += 32) {
        const float4* kp = (const float4*)(qkv + (size_t)(bT + j0 + jj) * (3 * DD) + DD + h * HD);
        float s = 0.0f;
#pragma unroll
        for (int d4 = 0; d4 < 16; ++d4) {
            float4 kvv = kp[d4];
            s += qv[d4].x * kvv.x + qv[d4].y * kvv.y + qv[d4].z * kvv.z + qv[d4].w * kvv.w;
        }
        sc[wid][jj] = s * 0.125f;        // 1/sqrt(64)
    }
    __syncthreads();

    // softmax over band (wave-local reduce)
    float m = -1e30f;
    for (int jj = lane; jj < nk; jj += 32) m = fmaxf(m, sc[wid][jj]);
#pragma unroll
    for (int o = 16; o > 0; o >>= 1) m = fmaxf(m, __shfl_xor(m, o, 32));
    float ssum = 0.0f;
    for (int jj = lane; jj < nk; jj += 32) {
        float p = __expf(sc[wid][jj] - m);
        sc[wid][jj] = p;
        ssum += p;
    }
#pragma unroll
    for (int o = 16; o > 0; o >>= 1) ssum += __shfl_xor(ssum, o, 32);
    float inv = 1.0f / ssum;
    __syncthreads();

    // pass 2: out = P @ V, 2 dims per lane
    int d0 = lane * 2;
    float a0 = 0.0f, a1 = 0.0f;
    for (int jj = 0; jj < nk; ++jj) {
        float p = sc[wid][jj];
        const float* vp = qkv + (size_t)(bT + j0 + jj) * (3 * DD) + 2 * DD + h * HD + d0;
        a0 += p * vp[0];
        a1 += p * vp[1];
    }
    u16* op = out_bf + (size_t)row * DD + h * HD + d0;
    op[0] = f2bf(a0 * inv);
    op[1] = f2bf(a1 * inv);
}

// ---------------- workspace layout (bytes) ----------------
static constexpr size_t OFF_XN   = 0;                                   // 8192*512*2
static constexpr size_t OFF_QKV  = OFF_XN   + (size_t)NTOK * DD * 2;    // 8192*1536*4
static constexpr size_t OFF_AOUT = OFF_QKV  + (size_t)NTOK * 3 * DD * 4;
static constexpr size_t OFF_X2   = OFF_AOUT + (size_t)NTOK * DD * 2;
static constexpr size_t OFF_H    = OFF_X2   + (size_t)NTOK * DD * 4;
static constexpr size_t OFF_WQKV = OFF_H    + (size_t)NTOK * FFN * 2;
static constexpr size_t OFF_WOUT = OFF_WQKV + (size_t)DD * 3 * DD * 2;
static constexpr size_t OFF_W1   = OFF_WOUT + (size_t)DD * DD * 2;
static constexpr size_t OFF_W2   = OFF_W1   + (size_t)DD * FFN * 2;

extern "C" void kernel_launch(void* const* d_in, const int* in_sizes, int n_in,
                              void* d_out, int out_size, void* d_ws, size_t ws_size,
                              hipStream_t stream) {
    (void)in_sizes; (void)n_in; (void)out_size; (void)ws_size;
    const float* x       = (const float*)d_in[0];
    const float* norm1_w = (const float*)d_in[1];
    const float* norm2_w = (const float*)d_in[2];
    const float* w_qkv   = (const float*)d_in[3];
    const float* w_out   = (const float*)d_in[4];
    const float* b_out   = (const float*)d_in[5];
    const float* w1      = (const float*)d_in[6];
    const float* b1      = (const float*)d_in[7];
    const float* w2      = (const float*)d_in[8];
    const float* b2      = (const float*)d_in[9];
    const int*   ctx     = (const int*)  d_in[10];

    char* ws = (char*)d_ws;
    u16*   xn_bf   = (u16*)  (ws + OFF_XN);
    float* qkv_f   = (float*)(ws + OFF_QKV);
    u16*   aout_bf = (u16*)  (ws + OFF_AOUT);
    float* x2_f    = (float*)(ws + OFF_X2);
    u16*   h_bf    = (u16*)  (ws + OFF_H);
    u16*   wqkv_bf = (u16*)  (ws + OFF_WQKV);
    u16*   wout_bf = (u16*)  (ws + OFF_WOUT);
    u16*   w1_bf   = (u16*)  (ws + OFF_W1);
    u16*   w2_bf   = (u16*)  (ws + OFF_W2);

    // pack weights to WMMA B-fragment layout (bf16)
    {
        int t;
        t = (DD/32)*((3*DD)/16)*32; pack_b_kernel<<<(t+255)/256,256,0,stream>>>(w_qkv, wqkv_bf, DD,  3*DD);
        t = (DD/32)*(DD/16)*32;     pack_b_kernel<<<(t+255)/256,256,0,stream>>>(w_out, wout_bf, DD,  DD);
        t = (DD/32)*(FFN/16)*32;    pack_b_kernel<<<(t+255)/256,256,0,stream>>>(w1,    w1_bf,   DD,  FFN);
        t = (FFN/32)*(DD/16)*32;    pack_b_kernel<<<(t+255)/256,256,0,stream>>>(w2,    w2_bf,   FFN, DD);
    }

    // --- attention sublayer ---
    rmsnorm_kernel<<<NTOK, 256, 0, stream>>>(x, norm1_w, xn_bf);

    gemm_bf16_wmma<<<dim3((3*DD)/64, NTOK/256), 256, 0, stream>>>(
        xn_bf, wqkv_bf, nullptr, nullptr, qkv_f, nullptr, NTOK, 3*DD, DD, 0);

    attn_band_kernel<<<(BB*HH*TT)/8, 256, 0, stream>>>(qkv_f, aout_bf, ctx);

    gemm_bf16_wmma<<<dim3(DD/64, NTOK/256), 256, 0, stream>>>(
        aout_bf, wout_bf, b_out, x, x2_f, nullptr, NTOK, DD, DD, 0);

    // --- FFN sublayer ---
    rmsnorm_kernel<<<NTOK, 256, 0, stream>>>(x2_f, norm2_w, xn_bf);

    gemm_bf16_wmma<<<dim3(FFN/64, NTOK/256), 256, 0, stream>>>(
        xn_bf, w1_bf, b1, nullptr, nullptr, h_bf, NTOK, FFN, DD, 1 /*gelu*/);

    gemm_bf16_wmma<<<dim3(DD/64, NTOK/256), 256, 0, stream>>>(
        h_bf, w2_bf, b2, x2_f, (float*)d_out, nullptr, NTOK, DD, FFN, 0);
}
